// Attention_26680336843340
// MI455X (gfx1250) — matrix operand order repaired and checked by reading, not compiled, not run on hardware
//
#include <hip/hip_runtime.h>
#include <hip/hip_bf16.h>

typedef __attribute__((ext_vector_type(16))) __bf16 v16bf;
typedef __attribute__((ext_vector_type(4)))  __bf16 v4bf;
typedef __attribute__((ext_vector_type(8)))  float  v8f;

#define HEADS 12
#define DHEAD 64
#define SEQ   2048
#define BATCH 4
#define DIM   768
#define INNER 768
#define QKVC  2304
#define SCALE 0.125f

// ---------------------------------------------------------------------------
// WMMA helper: D = A(16x32 bf16) * B(32x16 bf16) + C(16x16 f32)
// ---------------------------------------------------------------------------
__device__ __forceinline__ v8f wmma_bf16(v16bf a, v16bf b, v8f c) {
  return __builtin_amdgcn_wmma_f32_16x16x32_bf16(
      false, a, false, b, (short)0, c, false, false);
}

// A-fragment (16x32, MxK) load from a row-major row pointer (row = lane%16).
// ISA 7.12.2: element e -> K = (e<8 ? e : e+8) + 8*(lane>=16) (+k0)
__device__ __forceinline__ v16bf ldA(const __bf16* __restrict__ row, int lane, int k0) {
  const int g = (lane >> 4) & 1;
  v16bf a;
#pragma unroll
  for (int e = 0; e < 16; e += 2) {
    int k = ((e < 8) ? e : (e + 8)) + 8 * g + k0;
    a[e]     = row[k];
    a[e + 1] = row[k + 1];
  }
  return a;
}

// B-fragment (32x16, KxN): col = lane%16, element e -> K = (e<8 ? e : e+8) + 8*(lane>=16).
__device__ __forceinline__ v16bf ldB(const __bf16* __restrict__ p, int ld, int lane) {
  const int g = (lane >> 4) & 1, n = lane & 15;
  v16bf r;
#pragma unroll
  for (int e = 0; e < 16; ++e) {
    int k = ((e < 8) ? e : (e + 8)) + 8 * g;
    r[e] = p[k * ld + n];
  }
  return r;
}

// ---------------------------------------------------------------------------
// CDNA5 async global->LDS staging (ASYNCcnt-tracked, no VGPR bounce)
// ---------------------------------------------------------------------------
__device__ __forceinline__ unsigned lds_off(const void* p) {
  return (unsigned)(uintptr_t)p;   // LDS aperture: addr[31:0] is the DS offset
}
__device__ __forceinline__ void async_ld_b128(unsigned lds, const void* g) {
  asm volatile("global_load_async_to_lds_b128 %0, %1, off"
               :: "v"(lds), "v"(g) : "memory");
}
#define WAIT_ASYNC(n) asm volatile("s_wait_asynccnt " #n ::: "memory")

// ---------------------------------------------------------------------------
// 16-lane row reduction via DPP16 row_xmask (pure VALU, co-executes with WMMA)
// ---------------------------------------------------------------------------
template <int CTRL>
__device__ __forceinline__ float dpp_mov(float x) {
  return __builtin_bit_cast(float,
      __builtin_amdgcn_update_dpp(0, __builtin_bit_cast(int, x),
                                  CTRL, 0xf, 0xf, true));
}
__device__ __forceinline__ float redmax16(float v) {
  v = fmaxf(v, dpp_mov<0x161>(v));   // row_xmask:1
  v = fmaxf(v, dpp_mov<0x162>(v));   // row_xmask:2
  v = fmaxf(v, dpp_mov<0x164>(v));   // row_xmask:4
  v = fmaxf(v, dpp_mov<0x168>(v));   // row_xmask:8
  return v;
}
__device__ __forceinline__ float redsum16(float v) {
  v += dpp_mov<0x161>(v);
  v += dpp_mov<0x162>(v);
  v += dpp_mov<0x164>(v);
  v += dpp_mov<0x168>(v);
  return v;
}

// ---------------------------------------------------------------------------
// fp32 -> bf16 conversion (4 elems/thread, n is a multiple of 1024)
// ---------------------------------------------------------------------------
__global__ __launch_bounds__(256)
void f2bf_kernel(const float* __restrict__ in, __bf16* __restrict__ out, int n) {
  int i = (blockIdx.x * 256 + threadIdx.x) * 4;
  if (i < n) {
    float4 f = *(const float4*)(in + i);
    v4bf o;
    o[0] = (__bf16)f.x; o[1] = (__bf16)f.y; o[2] = (__bf16)f.z; o[3] = (__bf16)f.w;
    *(v4bf*)(out + i) = o;
  }
}

// ---------------------------------------------------------------------------
// Tiled bf16 WMMA GEMM, C tile 128x128, block = 256 threads (8 waves, 4x2),
// double-buffered LDS fed by async global->LDS loads.
// MODE 0: QKV projection epilogue -> scatter Q(scaled)/K^T/V (bf16)
// MODE 1: output projection epilogue -> f32 out + bias
// ---------------------------------------------------------------------------
template <int MODE>
__global__ __launch_bounds__(256)
void gemm_bf16_kernel(const __bf16* __restrict__ A, const __bf16* __restrict__ Bm,
                      int K, int lda, int ldb,
                      __bf16* __restrict__ qo, __bf16* __restrict__ ko,
                      __bf16* __restrict__ vo,
                      const float* __restrict__ bias, float* __restrict__ fo) {
  __shared__ __bf16 sA[2][128][40];   // 128 rows x 32 k  (pad 8)
  __shared__ __bf16 sB[2][32][136];   // 32 k   x 128 col (pad 8)
  const int tid  = threadIdx.x;
  const int lane = tid & 31;
  const int w    = tid >> 5;
  const int wm   = w >> 1;        // 0..3 : 32-row slab
  const int wn   = w & 1;         // 0..1 : 64-col slab
  const int row0 = blockIdx.y * 128;
  const int col0 = blockIdx.x * 128;

  // per-thread staging slots (2 x 16B for A, 2 x 16B for B)
  const int ra  = (tid >> 2),            ca  = (tid & 3) * 8;        // rows 0..63
  const int ra2 = ((tid + 256) >> 2),    ca2 = ((tid + 256) & 3) * 8; // rows 64..127
  const int rb  = (tid >> 4),            cb  = (tid & 15) * 8;       // k 0..15
  const int rb2 = ((tid + 256) >> 4),    cb2 = ((tid + 256) & 15) * 8; // k 16..31

  auto stage = [&](int buf, int kk) {
    async_ld_b128(lds_off(&sA[buf][ra][ca]),
                  &A[(size_t)(row0 + ra) * lda + kk + ca]);
    async_ld_b128(lds_off(&sA[buf][ra2][ca2]),
                  &A[(size_t)(row0 + ra2) * lda + kk + ca2]);
    async_ld_b128(lds_off(&sB[buf][rb][cb]),
                  &Bm[(size_t)(kk + rb) * ldb + col0 + cb]);
    async_ld_b128(lds_off(&sB[buf][rb2][cb2]),
                  &Bm[(size_t)(kk + rb2) * ldb + col0 + cb2]);
  };

  v8f acc[2][4];
#pragma unroll
  for (int i = 0; i < 2; ++i)
#pragma unroll
    for (int j = 0; j < 4; ++j) acc[i][j] = v8f{};

  stage(0, 0);
  const int steps = K >> 5;
  for (int it = 0; it < steps; ++it) {
    const int cur = it & 1;
    if (it + 1 < steps) {
      stage(cur ^ 1, (it + 1) * 32);
      WAIT_ASYNC(0x4);     // in-order: previous tile's 4 loads have landed
    } else {
      WAIT_ASYNC(0x0);
    }
    __syncthreads();

    v16bf af0 = ldA(&sA[cur][wm * 32 +      (lane & 15)][0], lane, 0);
    v16bf af1 = ldA(&sA[cur][wm * 32 + 16 + (lane & 15)][0], lane, 0);
#pragma unroll
    for (int t = 0; t < 4; ++t) {
      v16bf bf = ldB(&sB[cur][0][wn * 64 + t * 16], 136, lane);
      acc[0][t] = wmma_bf16(af0, bf, acc[0][t]);
      acc[1][t] = wmma_bf16(af1, bf, acc[1][t]);
    }
    __syncthreads();   // all waves done reading before buffer reuse
  }

  // C-fragment layout: lane L, reg r -> row = r + 8*(L>>4), col = L&15
  const int g = lane >> 4, nn = lane & 15;
#pragma unroll
  for (int sl = 0; sl < 2; ++sl)
#pragma unroll
    for (int t = 0; t < 4; ++t)
#pragma unroll
      for (int r = 0; r < 8; ++r) {
        int row = row0 + wm * 32 + sl * 16 + r + 8 * g;
        int col = col0 + wn * 64 + t * 16 + nn;
        float v = acc[sl][t][r];
        if (MODE == 0) {
          int b = row >> 11, n = row & 2047;
          int sec = (col >= 1536) ? 2 : ((col >= 768) ? 1 : 0);
          int cc = col - sec * 768;
          int h = cc >> 6, d = cc & 63;
          size_t bh = (size_t)(b * HEADS + h);
          if (sec == 0)       qo[(bh * SEQ + n) * DHEAD + d] = (__bf16)(v * SCALE);
          else if (sec == 1)  ko[(bh * DHEAD + d) * SEQ + n] = (__bf16)v;   // K transposed
          else                vo[(bh * SEQ + n) * DHEAD + d] = (__bf16)v;
        } else {
          fo[(size_t)row * DIM + col] = v + bias[col];
        }
      }
}

// ---------------------------------------------------------------------------
// Flash attention: grid (SEQ/64, HEADS, BATCH), block 128 = 4 waves.
// Each wave owns 16 query rows; block streams 32-key chunks through
// double-buffered LDS fed by async global->LDS loads.
// ---------------------------------------------------------------------------
__global__ __launch_bounds__(128)
void attn_kernel(const __bf16* __restrict__ Q, const __bf16* __restrict__ KT,
                 const __bf16* __restrict__ V, __bf16* __restrict__ O) {
  __shared__ __bf16 sKT[2][64][40];     // d x 32 keys (pad 8)
  __shared__ __bf16 sV[2][32][72];      // 32 keys x 64 d (pad 8)
  __shared__ __bf16 sP[4][16][40];      // per-wave P tile 16q x 32k (pad 8)

  const int tid = threadIdx.x, lane = tid & 31, w = tid >> 5;
  const int g = lane >> 4, nn = lane & 15;
  const size_t bh = (size_t)blockIdx.z * HEADS + blockIdx.y;
  const int q0 = blockIdx.x * 64 + w * 16;

  const __bf16* KTb = KT + bh * (size_t)DHEAD * SEQ;
  const __bf16* Vb  = V  + bh * (size_t)SEQ * DHEAD;

  // per-thread staging slots: sKT 64x32 = 256 16B chunks, sV 32x64 = 256 chunks
  const int ka  = (tid >> 2),         cka  = (tid & 3) * 8;        // d 0..31
  const int ka2 = ((tid + 128) >> 2), cka2 = ((tid + 128) & 3) * 8; // d 32..63
  const int va  = (tid >> 3),         cva  = (tid & 7) * 8;        // keys 0..15
  const int va2 = ((tid + 128) >> 3), cva2 = ((tid + 128) & 7) * 8; // keys 16..31

  auto stage = [&](int buf, int k0) {
    async_ld_b128(lds_off(&sKT[buf][ka][cka]),
                  &KTb[(size_t)ka * SEQ + k0 + cka]);
    async_ld_b128(lds_off(&sKT[buf][ka2][cka2]),
                  &KTb[(size_t)ka2 * SEQ + k0 + cka2]);
    async_ld_b128(lds_off(&sV[buf][va][cva]),
                  &Vb[(size_t)(k0 + va) * DHEAD + cva]);
    async_ld_b128(lds_off(&sV[buf][va2][cva2]),
                  &Vb[(size_t)(k0 + va2) * DHEAD + cva2]);
  };

  // Q fragments (Q already scaled by SCALE): A-frags for K-dim 0..31, 32..63
  const __bf16* qrow = Q + (bh * SEQ + q0 + nn) * DHEAD;
  v16bf qa0 = ldA(qrow, lane, 0);
  v16bf qa1 = ldA(qrow, lane, 32);

  v8f o0{}, o1{}, o2{}, o3{};
  float mrow[8], lrow[8];
#pragma unroll
  for (int r = 0; r < 8; ++r) { mrow[r] = -3.0e38f; lrow[r] = 0.0f; }

  stage(0, 0);
  for (int it = 0; it < SEQ / 32; ++it) {
    const int cur = it & 1;
    if (it + 1 < SEQ / 32) {
      stage(cur ^ 1, (it + 1) * 32);
      WAIT_ASYNC(0x4);
    } else {
      WAIT_ASYNC(0x0);
    }
    __syncthreads();

    // S = Q * K^T : two 16x16 tiles (keys k0..15, k0+16..31)
    v8f s0{}, s1{};
    s0 = wmma_bf16(qa0, ldB(&sKT[cur][0][0], 40, lane),   s0);
    s0 = wmma_bf16(qa1, ldB(&sKT[cur][32][0], 40, lane),  s0);
    s1 = wmma_bf16(qa0, ldB(&sKT[cur][0][16], 40, lane),  s1);
    s1 = wmma_bf16(qa1, ldB(&sKT[cur][32][16], 40, lane), s1);

    // online softmax; row stats live across the 16 lanes of each half-group
#pragma unroll
    for (int r = 0; r < 8; ++r) {
      float rm = redmax16(fmaxf(s0[r], s1[r]));
      float mnew  = fmaxf(mrow[r], rm);
      float alpha = __expf(mrow[r] - mnew);
      float p0 = __expf(s0[r] - mnew);
      float p1 = __expf(s1[r] - mnew);
      float rs = redsum16(p0 + p1);
      lrow[r] = lrow[r] * alpha + rs;
      mrow[r] = mnew;
      o0[r] *= alpha; o1[r] *= alpha; o2[r] *= alpha; o3[r] *= alpha;
      int q = r + 8 * g;                 // C-frag row for this reg/lane
      sP[w][q][nn]      = (__bf16)p0;    // bounce P through LDS (C->A relayout)
      sP[w][q][nn + 16] = (__bf16)p1;
    }
    __builtin_amdgcn_wave_barrier();     // per-wave LDS ops are HW-ordered

    // O += P(16x32) * V(32x64)
    v16bf pa = ldA(&sP[w][nn][0], lane, 0);
    o0 = wmma_bf16(pa, ldB(&sV[cur][0][0], 72, lane),  o0);
    o1 = wmma_bf16(pa, ldB(&sV[cur][0][16], 72, lane), o1);
    o2 = wmma_bf16(pa, ldB(&sV[cur][0][32], 72, lane), o2);
    o3 = wmma_bf16(pa, ldB(&sV[cur][0][48], 72, lane), o3);
    __syncthreads();
  }

  // write O back merged over heads: [B, N, H*64] bf16
  __bf16* Ob = O + ((size_t)blockIdx.z * SEQ + q0) * INNER + blockIdx.y * DHEAD;
#pragma unroll
  for (int r = 0; r < 8; ++r) {
    int q = r + 8 * g;
    float inv = 1.0f / lrow[r];
    __bf16* orow = Ob + (size_t)q * INNER;
    orow[nn]      = (__bf16)(o0[r] * inv);
    orow[16 + nn] = (__bf16)(o1[r] * inv);
    orow[32 + nn] = (__bf16)(o2[r] * inv);
    orow[48 + nn] = (__bf16)(o3[r] * inv);
  }
}

// ---------------------------------------------------------------------------
extern "C" void kernel_launch(void* const* d_in, const int* in_sizes, int n_in,
                              void* d_out, int out_size, void* d_ws, size_t ws_size,
                              hipStream_t stream) {
  const float* x     = (const float*)d_in[0];
  const float* w_qkv = (const float*)d_in[1];
  const float* w_out = (const float*)d_in[2];
  const float* b_out = (const float*)d_in[3];
  float* out = (float*)d_out;

  char* ws = (char*)d_ws;
  size_t off = 0;
  auto take = [&](size_t bytes) -> char* {
    char* p = ws + off;
    off += (bytes + 255) & ~(size_t)255;
    return p;
  };
  const size_t nx  = (size_t)BATCH * SEQ * DIM;
  const size_t nqk = (size_t)DIM * QKVC;
  const size_t nwo = (size_t)INNER * DIM;
  const size_t nh  = (size_t)BATCH * HEADS * SEQ * DHEAD;

  __bf16* Xbf  = (__bf16*)take(nx  * 2);
  __bf16* Wqkv = (__bf16*)take(nqk * 2);
  __bf16* Wout = (__bf16*)take(nwo * 2);
  __bf16* Qs   = (__bf16*)take(nh  * 2);
  __bf16* KT   = (__bf16*)take(nh  * 2);
  __bf16* Vb   = (__bf16*)take(nh  * 2);
  __bf16* Obf  = (__bf16*)take(nx  * 2);

  f2bf_kernel<<<(int)(nx  / 1024), 256, 0, stream>>>(x,     Xbf,  (int)nx);
  f2bf_kernel<<<(int)(nqk / 1024), 256, 0, stream>>>(w_qkv, Wqkv, (int)nqk);
  f2bf_kernel<<<(int)(nwo / 1024), 256, 0, stream>>>(w_out, Wout, (int)nwo);

  // QKV projection: [8192 x 768] * [768 x 2304]
  gemm_bf16_kernel<0><<<dim3(QKVC / 128, (BATCH * SEQ) / 128), 256, 0, stream>>>(
      Xbf, Wqkv, DIM, DIM, QKVC, Qs, KT, Vb, nullptr, nullptr);

  // flash attention
  attn_kernel<<<dim3(SEQ / 64, HEADS, BATCH), 128, 0, stream>>>(Qs, KT, Vb, Obf);

  // output projection: [8192 x 768] * [768 x 768] + bias -> f32
  gemm_bf16_kernel<1><<<dim3(DIM / 128, (BATCH * SEQ) / 128), 256, 0, stream>>>(
      Obf, Wout, INNER, INNER, DIM, nullptr, nullptr, nullptr, b_out, out);
}
